// GCN_420906795032
// MI455X (gfx1250) — matrix operand order repaired and checked by reading, not compiled, not run on hardware
//
#include <hip/hip_runtime.h>

// ---------------------------------------------------------------------------
// GCN (3x GCNConv, d=256) for MI455X / gfx1250, wave32.
//  - GEMM via v_wmma_f32_16x16x32_bf16, split-bf16 (hi/lo) fp32 emulation.
//    Each wave computes a 16x128 strip (1 M-tile x 8 N-tiles) so the A
//    fragment (relu + f32->bf16 hi/lo split) is built once and reused 8x.
//  - edge scatter via global_atomic_add_f32 (unsafeAtomicAdd); both 51 MB
//    node-feature buffers are L2-resident (192 MB L2).
// ---------------------------------------------------------------------------

typedef __bf16 bf16;
typedef __attribute__((ext_vector_type(16))) __bf16 v16bf;
typedef __attribute__((ext_vector_type(8)))  float  v8f;

#define DIM 256
#define NT  8      // N-tiles per wave (8 x 16 = 128 columns)

// ---------------- degree / norm ----------------

__global__ void k_init_deg(float* __restrict__ deg, int n) {
    int i = blockIdx.x * blockDim.x + threadIdx.x;
    if (i < n) deg[i] = 1.0f;                 // self-loop contributes 1
}

__global__ void k_edge_deg(const int* __restrict__ ei, float* __restrict__ deg, int E) {
    int e = blockIdx.x * blockDim.x + threadIdx.x;
    if (e < E) unsafeAtomicAdd(&deg[ei[E + e]], 1.0f);   // dst = ei[1][e]
}

__global__ void k_rsqrt(float* __restrict__ deg, int n) {
    int i = blockIdx.x * blockDim.x + threadIdx.x;
    if (i < n) deg[i] = rsqrtf(deg[i]);       // deg >= 1
}

// ---------------- W -> transposed bf16 hi/lo ----------------
// W is [in=256][out=256] row-major. Store Wt[n][k] so each lane's B-fragment
// is 16 contiguous bf16 (32 B).

__global__ void k_convert_w(const float* __restrict__ W,
                            bf16* __restrict__ WtHi, bf16* __restrict__ WtLo) {
    int idx = blockIdx.x * blockDim.x + threadIdx.x;   // 256*256 threads
    if (idx >= DIM * DIM) return;
    int k = idx >> 8;
    int n = idx & 255;
    float w  = W[idx];
    bf16 hi  = (bf16)w;
    bf16 lo  = (bf16)(w - (float)hi);
    WtHi[n * DIM + k] = hi;
    WtLo[n * DIM + k] = lo;
}

// ---------------- GEMM: H = act(X) @ W ----------------
// One wave32 per 16(M) x 128(N) strip; K=256 in 8 steps of 32.
// A layout (16x32): lane L -> row M=L&15;
//   lanes 0-15 hold K=kb..kb+15, lanes 16-31 hold K=kb+16..kb+31 (matches B).
// B layout (32x16): lane L -> column N=L&15; lanes 0-15 K=kb..kb+15,
//   lanes 16-31 K=kb+16..kb+31 (contiguous in Wt[n][k]).
// C/D: VGPR r, lane L -> (M = r + 8*(L>>4), N = L&15).

__global__ __launch_bounds__(256)
void k_gemm(const float* __restrict__ X,
            const bf16* __restrict__ WtHi, const bf16* __restrict__ WtLo,
            float* __restrict__ H, int mtiles, int relu) {
    int wave = (int)((blockIdx.x * blockDim.x + threadIdx.x) >> 5);
    int lane = threadIdx.x & 31;
    int tM    = wave >> 1;          // M tile
    int nhalf = wave & 1;           // which 128-column half
    if (tM >= mtiles) return;

    int half = lane >> 4;   // 0 or 1
    int mn   = lane & 15;

    const float* xrow = X + (size_t)(tM * 16 + mn) * DIM;
    // B fragment base for tile t: rows (nhalf*128 + t*16 + mn) of Wt
    size_t boff = (size_t)(nhalf * 128 + mn) * DIM + 16 * half;
    const bf16* bh_base = WtHi + boff;
    const bf16* bl_base = WtLo + boff;

    v8f acc[NT];
#pragma unroll
    for (int t = 0; t < NT; ++t) acc[t] = (v8f){};

#pragma unroll
    for (int kb = 0; kb < DIM; kb += 32) {
        // ---- build A fragment once (relu + hi/lo bf16 split) ----
        v16bf ah, al;
        int k0 = kb + 16 * half;
#pragma unroll
        for (int e = 0; e < 8; ++e) {
            float v0 = xrow[k0 + e];
            float v1 = xrow[k0 + 8 + e];
            if (relu) { v0 = fmaxf(v0, 0.0f); v1 = fmaxf(v1, 0.0f); }
            bf16 h0 = (bf16)v0;
            bf16 h1 = (bf16)v1;
            ah[e]     = h0;
            ah[e + 8] = h1;
            al[e]     = (bf16)(v0 - (float)h0);
            al[e + 8] = (bf16)(v1 - (float)h1);
        }
        // ---- reuse A across 8 B tiles ----
#pragma unroll
        for (int t = 0; t < NT; ++t) {
            const bf16* bhrow = bh_base + (size_t)t * 16 * DIM + kb;
            const bf16* blrow = bl_base + (size_t)t * 16 * DIM + kb;
            v16bf bh, bl;
#pragma unroll
            for (int e = 0; e < 16; ++e) { bh[e] = bhrow[e]; bl[e] = blrow[e]; }
            // split-bf16 fp32 emulation: hi*hi + hi*lo + lo*hi
            acc[t] = __builtin_amdgcn_wmma_f32_16x16x32_bf16(false, ah, false, bh, (short)0, acc[t], false, false);
            acc[t] = __builtin_amdgcn_wmma_f32_16x16x32_bf16(false, ah, false, bl, (short)0, acc[t], false, false);
            acc[t] = __builtin_amdgcn_wmma_f32_16x16x32_bf16(false, al, false, bh, (short)0, acc[t], false, false);
        }
    }

    int mbase = tM * 16 + 8 * half;
#pragma unroll
    for (int t = 0; t < NT; ++t) {
        int n = nhalf * 128 + t * 16 + mn;
#pragma unroll
        for (int r = 0; r < 8; ++r)
            H[(size_t)(mbase + r) * DIM + n] = acc[t][r];
    }
}

// ---------------- agg init: bias + self-loop ----------------

__global__ void k_init_agg(const float* __restrict__ H, const float* __restrict__ dinv,
                           const float* __restrict__ bias, float* __restrict__ A, int total) {
    int i = blockIdx.x * blockDim.x + threadIdx.x;
    if (i < total) {
        int node = i >> 8;
        int f    = i & 255;
        float di = dinv[node];
        A[i] = bias[f] + di * di * H[i];
    }
}

// ---------------- edge scatter: one wave32 per edge ----------------

__global__ __launch_bounds__(256)
void k_edge_agg(const int* __restrict__ ei, const float* __restrict__ dinv,
                const float* __restrict__ H, float* __restrict__ A, int E) {
    long long gid = (long long)blockIdx.x * blockDim.x + threadIdx.x;
    int e    = (int)(gid >> 5);
    int lane = (int)(gid & 31);
    if (e >= E) return;
    int s = ei[e];
    int d = ei[E + e];
    float nrm = dinv[s] * dinv[d];
    const float4* hv = (const float4*)(H + (size_t)s * DIM);
    float4 a = hv[lane * 2];
    float4 b = hv[lane * 2 + 1];
    float* out = A + (size_t)d * DIM + lane * 8;
    unsafeAtomicAdd(out + 0, a.x * nrm);
    unsafeAtomicAdd(out + 1, a.y * nrm);
    unsafeAtomicAdd(out + 2, a.z * nrm);
    unsafeAtomicAdd(out + 3, a.w * nrm);
    unsafeAtomicAdd(out + 4, b.x * nrm);
    unsafeAtomicAdd(out + 5, b.y * nrm);
    unsafeAtomicAdd(out + 6, b.z * nrm);
    unsafeAtomicAdd(out + 7, b.w * nrm);
}

// ---------------- driver ----------------

extern "C" void kernel_launch(void* const* d_in, const int* in_sizes, int n_in,
                              void* d_out, int out_size, void* d_ws, size_t ws_size,
                              hipStream_t stream) {
    const float* x    = (const float*)d_in[0];
    const float* W[3] = { (const float*)d_in[1], (const float*)d_in[3], (const float*)d_in[5] };
    const float* B[3] = { (const float*)d_in[2], (const float*)d_in[4], (const float*)d_in[6] };
    const int*   ei   = (const int*)d_in[7];

    const int Nn = in_sizes[0] / DIM;       // 50000
    const int E  = in_sizes[7] / 2;         // 1.6M

    // workspace layout
    char* ws = (char*)d_ws;
    float* dinv = (float*)ws;                                 // Nn floats
    bf16*  WtHi = (bf16*)(ws + 262144);                       // 128 KB
    bf16*  WtLo = (bf16*)(ws + 262144 + 131072);              // 128 KB
    float* Hbuf = (float*)(ws + (1 << 20));                   // Nn*256 floats (~51.2 MB)
    float* out  = (float*)d_out;                              // agg / activation ping buffer

    const int T = 256;

    // degrees -> dinv
    k_init_deg<<<(Nn + T - 1) / T, T, 0, stream>>>(dinv, Nn);
    k_edge_deg<<<(E + T - 1) / T, T, 0, stream>>>(ei, dinv, E);
    k_rsqrt  <<<(Nn + T - 1) / T, T, 0, stream>>>(dinv, Nn);

    const int mtiles      = (Nn + 15) / 16;
    const int waves       = mtiles * 2;              // 2 N-strips of 128 cols
    const int gemm_blocks = (waves + 7) / 8;         // 8 waves per 256-thread block
    const int total       = Nn * DIM;
    const long long ethreads = (long long)E * 32;

    for (int l = 0; l < 3; ++l) {
        k_convert_w<<<(DIM * DIM + T - 1) / T, T, 0, stream>>>(W[l], WtHi, WtLo);
        const float* Xin = (l == 0) ? x : out;       // relu fused into A-load for l>0
        k_gemm<<<gemm_blocks, T, 0, stream>>>(Xin, WtHi, WtLo, Hbuf, mtiles, l > 0 ? 1 : 0);
        k_init_agg<<<(total + T - 1) / T, T, 0, stream>>>(Hbuf, dinv, B[l], out, total);
        k_edge_agg<<<(int)((ethreads + T - 1) / T), T, 0, stream>>>(ei, dinv, Hbuf, out, E);
    }
}
